// NeRFRenderer_6897717477569
// MI455X (gfx1250) — compile-verified, matrix operand order checked by reading, not run on hardware
//
#include <hip/hip_runtime.h>
#include <hip/hip_bf16.h>

#define N_RAYS 8192
#define S_STEPS 128
#define EPSF 0.005f

typedef __attribute__((ext_vector_type(16))) _Float16 v16h;
typedef __attribute__((ext_vector_type(8)))  float    v8f;

union HFrag { v16h h; unsigned u[8]; };

// K index held by VGPR pair v (0..7) for the 16-bit 16x32 A / 32x16 B layouts:
// lanes 0-15: K = 0..7 (v0..3), 16..23 (v4..7); lanes 16-31: +8.
__device__ __forceinline__ int kbase_of(int v, int half) {
    return ((v < 4) ? (2 * v) : (16 + 2 * (v - 4))) + 8 * half;
}

// A fragment (16x32 f16) from per-wave LDS activation tile act[16][72]
__device__ __forceinline__ v16h load_a_lds(const _Float16* act, int lane, int kk) {
    HFrag f;
    const int m = lane & 15, half = lane >> 4;
    const _Float16* row = act + m * 72 + 32 * kk;
#pragma unroll
    for (int v = 0; v < 8; ++v)
        f.u[v] = *(const unsigned*)(row + kbase_of(v, half));
    return f.h;
}

// B fragment (32x16 f16) from transposed weights Wt[n][64] in global (L2-resident)
__device__ __forceinline__ v16h load_b_glb(const _Float16* Wt, int fblk, int lane, int kk) {
    HFrag f;
    const int half = lane >> 4;
    const _Float16* row = Wt + (16 * fblk + (lane & 15)) * 64 + 32 * kk;
#pragma unroll
    for (int v = 0; v < 8; ++v)
        f.u[v] = *(const unsigned*)(row + kbase_of(v, half));
    return f.h;
}

__device__ __forceinline__ v8f wmma32(v16h a, v16h b, v8f c) {
    return __builtin_amdgcn_wmma_f32_16x16x32_f16(false, a, false, b, (short)0, c, false, false);
}

__device__ __forceinline__ float sigmoidf(float x) { return 1.f / (1.f + expf(-x)); }

// 64 -> 64 dense + ReLU: 8 WMMAs with register-resident B fragments
__device__ __forceinline__ void dense64_relu_r(const _Float16* actIn, _Float16* actOut,
                                               const v16h (&B)[4][2], const float* bias,
                                               int lane) {
    v16h a0 = load_a_lds(actIn, lane, 0);
    v16h a1 = load_a_lds(actIn, lane, 1);
    const int n = lane & 15, half = lane >> 4;
#pragma unroll
    for (int f = 0; f < 4; ++f) {
        v8f acc;
        float bn = bias[16 * f + n];
#pragma unroll
        for (int r = 0; r < 8; ++r) acc[r] = bn;
        acc = wmma32(a0, B[f][0], acc);
        acc = wmma32(a1, B[f][1], acc);
#pragma unroll
        for (int r = 0; r < 8; ++r)  // D element (M=r+8*half, N=16f+n) -> act[m][k]
            actOut[(r + 8 * half) * 72 + 16 * f + n] = (_Float16)fmaxf(acc[r], 0.f);
    }
}

// 64 -> 16 dense (bias preloaded into acc): 2 WMMAs, register-resident B
__device__ __forceinline__ v8f dense16_r(const _Float16* actIn, v16h b0, v16h b1,
                                         v8f acc, int lane) {
    v16h a0 = load_a_lds(actIn, lane, 0);
    v16h a1 = load_a_lds(actIn, lane, 1);
    acc = wmma32(a0, b0, acc);
    acc = wmma32(a1, b1, acc);
    return acc;
}

// --------- prep: transpose weights to f16 [n][64] fragments ----------
__global__ __launch_bounds__(256) void neus_prep(const float* w1s, const float* w2s,
                                                 const float* wc1, const float* wc2,
                                                 _Float16* W1tS, _Float16* W2tS,
                                                 _Float16* W1tC, _Float16* W2tC) {
    int i = blockIdx.x * blockDim.x + threadIdx.x;
    if (i < 4096) {
        int n = i >> 6, k = i & 63;
        W1tS[i] = (_Float16)w1s[k * 64 + n];
        W1tC[i] = (_Float16)wc1[k * 64 + n];
    }
    if (i < 1024) {
        int n = i >> 6, k = i & 63;
        W2tS[i] = (_Float16)w2s[k * 16 + n];
        W2tC[i] = (n < 3) ? (_Float16)wc2[k * 3 + n] : (_Float16)0.f;
    }
}

// --------- main: 1 block = 1 ray; 4 waves x 2 passes = 8 tiles of 16 samples ----------
__global__ __launch_bounds__(128) void neus_main(
    const float* __restrict__ rays_o, const float* __restrict__ rays_d,
    const float* __restrict__ b0s, const float* __restrict__ b1s,
    const float* __restrict__ b2s, const float* __restrict__ w0s,
    const float* __restrict__ bc0, const float* __restrict__ bc1,
    const float* __restrict__ bc2, const float* __restrict__ wc0,
    const float* __restrict__ var_s,
    const _Float16* __restrict__ W1tS, const _Float16* __restrict__ W2tS,
    const _Float16* __restrict__ W1tC, const _Float16* __restrict__ W2tC,
    float* __restrict__ gesums, float* __restrict__ out) {
    __shared__ _Float16 s_actA[4][16 * 72];  // activation ping (per wave)
    __shared__ _Float16 s_actB[4][16 * 72];  // activation pong (per wave)
    __shared__ float    s_sdfcol[4][7][16];  // sdf (col 0) per finite-diff variant
    __shared__ float    s_feat[4][16][16];   // full sdf-net output of base variant
    __shared__ float    s_x[4][16][4];       // clipped sample positions + |pt|
    __shared__ float    s_stage[S_STEPS][10];// c0..2, n0..2, sdf, truecos, |grad|, |pt|
    __shared__ float    s_hdr[8];            // ro xyz, rd xyz, near, far
    __shared__ float    s_w0[192], s_b0[64]; // sdf layer-0 weights/bias
    __shared__ float    s_wc0[1536], s_bc0[64];  // color layer-0 weights/bias
    __shared__ float    s_b1[64], s_bc1[64], s_b2[16], s_bc2[4];

    const int ray  = blockIdx.x;
    const int tid  = threadIdx.x;
    const int wv   = tid >> 5;
    const int lane = tid & 31;

    // ---- stage small weights to LDS (one pass; L2-resident across all blocks) ----
    for (int i = tid; i < 1536; i += 128) s_wc0[i] = wc0[i];
    for (int i = tid; i < 192; i += 128) s_w0[i] = w0s[i];
    if (tid < 64) {
        s_b0[tid] = b0s[tid];
        s_b1[tid] = b1s[tid];
        s_bc0[tid] = bc0[tid];
        s_bc1[tid] = bc1[tid];
        if (tid < 16) s_b2[tid] = b2s[tid];
        if (tid < 3) s_bc2[tid] = bc2[tid];
    }
    if (tid == 0) {
        __builtin_prefetch(W1tS, 0, 3);
        __builtin_prefetch(W2tS, 0, 3);
        __builtin_prefetch(W1tC, 0, 3);
        __builtin_prefetch(W2tC, 0, 3);
        float nearv = -1e30f, farv = 1e30f;
        for (int j = 0; j < 3; ++j) {
            float o = rays_o[ray * 3 + j];
            float d = rays_d[ray * 3 + j];
            float dn = d + 1e-15f;
            float t0 = (-1.f - o) / dn, t1 = (1.f - o) / dn;
            nearv = fmaxf(nearv, fminf(t0, t1));
            farv  = fminf(farv,  fmaxf(t0, t1));
            s_hdr[j] = o;
            s_hdr[3 + j] = d;
        }
        s_hdr[6] = fmaxf(nearv, 0.05f);
        s_hdr[7] = farv;
    }
    __syncthreads();
    const float nearv = s_hdr[6], farv = s_hdr[7];
    const float dz = (farv - nearv) * (1.f / (S_STEPS - 1));

    // ---- SDF network B fragments: loop-invariant, keep in registers (80 VGPRs) ----
    v16h Bs1[4][2];
#pragma unroll
    for (int f = 0; f < 4; ++f) {
        Bs1[f][0] = load_b_glb(W1tS, f, lane, 0);
        Bs1[f][1] = load_b_glb(W1tS, f, lane, 1);
    }
    v16h Bs2_0 = load_b_glb(W2tS, 0, lane, 0);
    v16h Bs2_1 = load_b_glb(W2tS, 0, lane, 1);

    // ---- per-lane layer-0 weight columns (each lane owns k = lane, lane+32) ----
    float w0c[2][3], b0c[2];
#pragma unroll
    for (int kc = 0; kc < 2; ++kc) {
        const int k = lane + 32 * kc;
#pragma unroll
        for (int j = 0; j < 3; ++j) w0c[kc][j] = s_w0[j * 64 + k];
        b0c[kc] = s_b0[k];
    }

    for (int pass = 0; pass < 2; ++pass) {
        const int tile = wv + 4 * pass;
        const int s0 = tile * 16;
        _Float16* aA = s_actA[wv];
        _Float16* aB = s_actB[wv];

        // ---- sample positions (clipped) ----
        if (lane < 16) {
            int s = s0 + lane;
            float zv = nearv + dz * (float)s;
            float zm = (s < S_STEPS - 1) ? (zv + 0.5f * dz) : zv;
            float pn2 = 0.f;
            for (int j = 0; j < 3; ++j) {
                float xj = fminf(fmaxf(s_hdr[j] + s_hdr[3 + j] * zm, -1.f), 1.f);
                s_x[wv][lane][j] = xj;
                pn2 += xj * xj;
            }
            s_x[wv][lane][3] = sqrtf(pn2);
        }

        // ---- SDF layer0 pre-activation (linear in x) -> registers, 2 cols/lane ----
        float ubase[2][16];
#pragma unroll
        for (int kc = 0; kc < 2; ++kc) {
#pragma unroll
            for (int m = 0; m < 16; ++m) {
                float acc = b0c[kc];
                acc += s_x[wv][m][0] * w0c[kc][0];
                acc += s_x[wv][m][1] * w0c[kc][1];
                acc += s_x[wv][m][2] * w0c[kc][2];
                ubase[kc][m] = acc;
            }
        }

        // ---- 7 SDF evals: base + (+/-eps) per axis; layer0 shift = +/-eps*W0[axis] ----
        for (int v = 0; v < 7; ++v) {
            const int axis = (v > 0) ? ((v - 1) >> 1) : 0;
            const float se = (((v - 1) & 1) == 0) ? EPSF : -EPSF;
#pragma unroll
            for (int kc = 0; kc < 2; ++kc) {
                const int k = lane + 32 * kc;
                const float dv = (v > 0) ? se * s_w0[axis * 64 + k] : 0.f;
#pragma unroll
                for (int m = 0; m < 16; ++m)
                    aA[m * 72 + k] = (_Float16)fmaxf(ubase[kc][m] + dv, 0.f);
            }
            dense64_relu_r(aA, aB, Bs1, s_b1, lane);
            v8f acc;
            float bn = s_b2[lane & 15];
#pragma unroll
            for (int r = 0; r < 8; ++r) acc[r] = bn;
            acc = dense16_r(aB, Bs2_0, Bs2_1, acc, lane);
            const int half = lane >> 4;
            if ((lane & 15) == 0) {
#pragma unroll
                for (int r = 0; r < 8; ++r) s_sdfcol[wv][v][r + 8 * half] = acc[r];
            }
            if (v == 0) {
#pragma unroll
                for (int r = 0; r < 8; ++r) s_feat[wv][r + 8 * half][lane & 15] = acc[r];
            }
        }

        // ---- gradient / normal / truecos per sample ----
        if (lane < 16) {
            const int m = lane;
            float g[3], gn2 = 0.f;
            for (int j = 0; j < 3; ++j) {
                g[j] = (s_sdfcol[wv][1 + 2 * j][m] - s_sdfcol[wv][2 + 2 * j][m]) * (1.f / (2.f * EPSF));
                gn2 += g[j] * g[j];
            }
            float gn = sqrtf(gn2);
            float inv = 1.f / (1e-5f + gn);
            float tc = 0.f;
            int s = s0 + m;
            for (int j = 0; j < 3; ++j) {
                float nj = g[j] * inv;
                s_stage[s][3 + j] = nj;
                tc += nj * s_hdr[3 + j];
            }
            s_stage[s][6] = s_sdfcol[wv][0][m];
            s_stage[s][7] = tc;
            s_stage[s][8] = gn;
            s_stage[s][9] = s_x[wv][m][3];
        }

        // ---- color layer0 (K=24: x, d, normal, feat), weights register-resident ----
#pragma unroll
        for (int kc = 0; kc < 2; ++kc) {
            const int k = lane + 32 * kc;
            float wcol[24];
#pragma unroll
            for (int j = 0; j < 24; ++j) wcol[j] = s_wc0[j * 64 + k];
            const float base = s_bc0[k] + s_hdr[3] * wcol[3] + s_hdr[4] * wcol[4] +
                               s_hdr[5] * wcol[5];
            for (int m = 0; m < 16; ++m) {
                int s = s0 + m;
                float acc = base;
#pragma unroll
                for (int j = 0; j < 3; ++j)  acc += s_x[wv][m][j]     * wcol[j];
#pragma unroll
                for (int j = 0; j < 3; ++j)  acc += s_stage[s][3 + j] * wcol[6 + j];
#pragma unroll
                for (int j = 0; j < 15; ++j) acc += s_feat[wv][m][1 + j] * wcol[9 + j];
                aA[m * 72 + k] = (_Float16)fmaxf(acc, 0.f);
            }
        }
        {
            // color hidden + output layers (B loaded per pass; cheap vs. variant loop)
            v16h Bc1[4][2];
#pragma unroll
            for (int f = 0; f < 4; ++f) {
                Bc1[f][0] = load_b_glb(W1tC, f, lane, 0);
                Bc1[f][1] = load_b_glb(W1tC, f, lane, 1);
            }
            dense64_relu_r(aA, aB, Bc1, s_bc1, lane);
            v8f acc;
            const int n = lane & 15, half = lane >> 4;
            float bn = (n < 3) ? s_bc2[n] : 0.f;
#pragma unroll
            for (int r = 0; r < 8; ++r) acc[r] = bn;
            acc = dense16_r(aB, load_b_glb(W2tC, 0, lane, 0),
                            load_b_glb(W2tC, 0, lane, 1), acc, lane);
            if (n < 3) {
#pragma unroll
                for (int r = 0; r < 8; ++r)
                    s_stage[s0 + r + 8 * half][n] = sigmoidf(acc[r]);
            }
        }
    }
    __syncthreads();

    // ---- per-ray render: sequential transmittance scan ----
    if (tid == 0) {
        const float inv_s = expf(10.f * var_s[0]);
        const float dlast = (farv - nearv) * (1.f / S_STEPS);
        float T = 1.f, img0 = 0.f, img1 = 0.f, img2 = 0.f;
        float nm0 = 0.f, nm1 = 0.f, nm2 = 0.f;
        float depth = 0.f, wsum = 0.f, geS = 0.f, rlS = 0.f;
        for (int s = 0; s < S_STEPS; ++s) {
            const float* st = s_stage[s];
            float delta = (s < S_STEPS - 1) ? dz : dlast;
            float t = -100.f * st[7];
            float sp = (t > 20.f) ? t : log1pf(expf(t));
            float iter_cos = -(sp * 0.01f);
            float e = iter_cos * delta * 0.5f;
            float pc = sigmoidf((st[6] - e) * inv_s);
            float nc = sigmoidf((st[6] + e) * inv_s);
            float alpha = fminf(fmaxf((pc - nc + 1e-5f) / (pc + 1e-5f), 0.f), 1.f);
            float w = alpha * T;
            T *= (1.f - alpha + 1e-7f);
            img0 += st[0] * w; img1 += st[1] * w; img2 += st[2] * w;
            nm0 += st[3] * w;  nm1 += st[4] * w;  nm2 += st[5] * w;
            depth += w * ((float)s * (1.f / (S_STEPS - 1)));
            wsum += w;
            float relax = (st[9] < 1.2f) ? 1.f : 0.f;
            geS += relax * (st[8] - 1.f) * (st[8] - 1.f);
            rlS += relax;
        }
        float bg = 1.f - wsum;  // bg_color == 1
        out[ray] = depth;
        out[N_RAYS + ray * 3 + 0] = img0 + bg;
        out[N_RAYS + ray * 3 + 1] = img1 + bg;
        out[N_RAYS + ray * 3 + 2] = img2 + bg;
        out[N_RAYS * 4 + ray * 3 + 0] = nm0;
        out[N_RAYS * 4 + ray * 3 + 1] = nm1;
        out[N_RAYS * 4 + ray * 3 + 2] = nm2;
        gesums[2 * ray] = geS;
        gesums[2 * ray + 1] = rlS;
    }
}

// --------- deterministic gradient_error reduction ----------
__global__ __launch_bounds__(256) void neus_finalize(const float* __restrict__ gesums,
                                                     float* __restrict__ out) {
    __shared__ float sge[256], srl[256];
    float ge = 0.f, rl = 0.f;
    for (int i = threadIdx.x; i < N_RAYS; i += 256) {
        ge += gesums[2 * i];
        rl += gesums[2 * i + 1];
    }
    sge[threadIdx.x] = ge;
    srl[threadIdx.x] = rl;
    __syncthreads();
    for (int s = 128; s > 0; s >>= 1) {
        if (threadIdx.x < (unsigned)s) {
            sge[threadIdx.x] += sge[threadIdx.x + s];
            srl[threadIdx.x] += srl[threadIdx.x + s];
        }
        __syncthreads();
    }
    if (threadIdx.x == 0) {
        out[N_RAYS * 7]     = sge[0] / (srl[0] + 1e-5f);
        out[N_RAYS * 7 + 1] = 0.f;
    }
}

extern "C" void kernel_launch(void* const* d_in, const int* in_sizes, int n_in,
                              void* d_out, int out_size, void* d_ws, size_t ws_size,
                              hipStream_t stream) {
    (void)in_sizes; (void)n_in; (void)out_size; (void)ws_size;
    // pytree flatten order: rays_o, rays_d, params{color.b0..b2,w0..w2, sdf.b0..b2,w0..w2,
    // variance_s}, num_steps, bound, upsample_steps, bg_color
    const float* rays_o = (const float*)d_in[0];
    const float* rays_d = (const float*)d_in[1];
    const float* bc0 = (const float*)d_in[2];
    const float* bc1 = (const float*)d_in[3];
    const float* bc2 = (const float*)d_in[4];
    const float* wc0 = (const float*)d_in[5];
    const float* wc1 = (const float*)d_in[6];
    const float* wc2 = (const float*)d_in[7];
    const float* b0s = (const float*)d_in[8];
    const float* b1s = (const float*)d_in[9];
    const float* b2s = (const float*)d_in[10];
    const float* w0s = (const float*)d_in[11];
    const float* w1s = (const float*)d_in[12];
    const float* w2s = (const float*)d_in[13];
    const float* vars = (const float*)d_in[14];

    _Float16* W1tS = (_Float16*)d_ws;          // 4096 f16
    _Float16* W2tS = W1tS + 4096;              // 1024 f16
    _Float16* W1tC = W2tS + 1024;              // 4096 f16
    _Float16* W2tC = W1tC + 4096;              // 1024 f16 (cols >=3 zero)
    float* gesums = (float*)((char*)d_ws + 20480);  // 2 * N_RAYS floats
    float* out = (float*)d_out;

    neus_prep<<<16, 256, 0, stream>>>(w1s, w2s, wc1, wc2, W1tS, W2tS, W1tC, W2tC);
    neus_main<<<N_RAYS, 128, 0, stream>>>(rays_o, rays_d, b0s, b1s, b2s, w0s,
                                          bc0, bc1, bc2, wc0, vars,
                                          W1tS, W2tS, W1tC, W2tC, gesums, out);
    neus_finalize<<<1, 256, 0, stream>>>(gesums, out);
}
